// DecoderWithAttention_10084583211229
// MI455X (gfx1250) — compile-verified, matrix operand order checked by reading, not run on hardware
//
#include <hip/hip_runtime.h>
#include <hip/hip_bf16.h>
#include <math.h>

// Problem dims (fixed by the reference)
#define BB   2048
#define SS   64
#define EE   256
#define DD   512
#define NIN  32
#define TT   64
#define HIST 31
#define G4D  (4*DD)

typedef __attribute__((ext_vector_type(16))) __bf16 bf16x16;
typedef __attribute__((ext_vector_type(8)))  __bf16 bf16x8;
typedef __attribute__((ext_vector_type(8)))  float  floatx8;

// ---------------------------------------------------------------------------
// WMMA fragment loader: one lane holds 16 bf16 of a 16x32 tile row.
// Layout (ISA 7.12.2, 16-bit A 16x32): lane%16 = M; half = lane/16;
// elements 0..7  -> K = k0 + half*8 .. +7
// elements 8..15 -> K = k0 + 16 + half*8 .. +7
// ---------------------------------------------------------------------------
__device__ __forceinline__ bf16x16 load_frag(const __bf16* p) {
    bf16x8 lo = *(const bf16x8*)(p);
    bf16x8 hi = *(const bf16x8*)(p + 16);
    return __builtin_shufflevector(lo, hi, 0,1,2,3,4,5,6,7,8,9,10,11,12,13,14,15);
}

#define WMMA_BF16(a, b, c) __builtin_amdgcn_wmma_f32_16x16x32_bf16( \
    false, (a), false, (b), (short)0, (c), false, false)

// Software-pipelined fragment accumulator: preloads next k-step's fragments
// before issuing the current step's 4 WMMAs, so v_wmma overlaps the loads.
// jstride = element offset between the 4 B tiles (16*K for consecutive
// n-tiles; DD*K for the i/f/g/o gate-split tiles).
__device__ __forceinline__ void gemm_acc(const __bf16* __restrict__ A,
                                         const __bf16* __restrict__ Bt,
                                         int K, size_t jstride,
                                         int m0, int n0, int lm, int half,
                                         floatx8* acc) {
    const __bf16* arow = A  + (size_t)(m0 + lm) * K + half * 8;
    const __bf16* brow = Bt + (size_t)(n0 + lm) * K + half * 8;
    bf16x16 a  = load_frag(arow);
    bf16x16 b0 = load_frag(brow);
    bf16x16 b1 = load_frag(brow + jstride);
    bf16x16 b2 = load_frag(brow + 2 * jstride);
    bf16x16 b3 = load_frag(brow + 3 * jstride);
    for (int k0 = 0; k0 + 32 < K; k0 += 32) {
        const int kn = k0 + 32;
        bf16x16 an  = load_frag(arow + kn);
        bf16x16 bn0 = load_frag(brow + kn);
        bf16x16 bn1 = load_frag(brow + jstride + kn);
        bf16x16 bn2 = load_frag(brow + 2 * jstride + kn);
        bf16x16 bn3 = load_frag(brow + 3 * jstride + kn);
        acc[0] = WMMA_BF16(a, b0, acc[0]);
        acc[1] = WMMA_BF16(a, b1, acc[1]);
        acc[2] = WMMA_BF16(a, b2, acc[2]);
        acc[3] = WMMA_BF16(a, b3, acc[3]);
        a = an; b0 = bn0; b1 = bn1; b2 = bn2; b3 = bn3;
    }
    acc[0] = WMMA_BF16(a, b0, acc[0]);
    acc[1] = WMMA_BF16(a, b1, acc[1]);
    acc[2] = WMMA_BF16(a, b2, acc[2]);
    acc[3] = WMMA_BF16(a, b3, acc[3]);
}

// ---------------------------------------------------------------------------
// Generic GEMM: C[M,N] = A1@B1t^T (+bias); opt tanh; opt bf16 copy.
// Block = 256 threads = 8 waves; wave tile 16x64; block tile 128x64.
// ---------------------------------------------------------------------------
__global__ __launch_bounds__(256) void gemm_bf16_wmma(
    const __bf16* __restrict__ A1, const __bf16* __restrict__ B1, int K1,
    const float* __restrict__ bias,
    float* __restrict__ C, __bf16* __restrict__ Cbf,
    int N, int act)
{
    const int lane = threadIdx.x & 31;
    const int wave = threadIdx.x >> 5;
    const int lm   = lane & 15;
    const int half = lane >> 4;
    const int m0   = blockIdx.y * 128 + wave * 16;
    const int n0   = blockIdx.x * 64;

    floatx8 acc[4] = {};
    gemm_acc(A1, B1, K1, (size_t)16 * K1, m0, n0, lm, half, acc);

#pragma unroll
    for (int j = 0; j < 4; ++j) {
        const int n = n0 + j * 16 + lm;
        const float bv = bias ? bias[n] : 0.0f;
#pragma unroll
        for (int r = 0; r < 8; ++r) {
            // D layout: VGPR r -> M = r + 8*half; lane%16 -> N
            const int m = m0 + half * 8 + r;
            float val = acc[j][r] + bv;
            if (act == 1) val = tanhf(val);
            C[(size_t)m * N + n] = val;
            if (Cbf) Cbf[(size_t)m * N + n] = (__bf16)val;
        }
    }
}

__device__ __forceinline__ float sigm(float x) { return 1.0f / (1.0f + expf(-x)); }

// ---------------------------------------------------------------------------
// Fused LSTM step: gates GEMM + nonlinearity + cell update in one kernel.
// acc[j] = gate-j 16x16 tile at (m0,n0): rows j*DD+n of Whh/Wih (jstride=DD*K).
// gates = h@Whh^T + x_t@Wih^T + (b_ih+b_hh); c,h updated in epilogue.
// hbf_in and hbf_out are distinct buffers (ping-pong) to avoid RAW races.
// Block = 8 waves -> 128(M) x 16(N within D); grid = (DD/16, BB/128).
// ---------------------------------------------------------------------------
__global__ __launch_bounds__(256) void lstm_step_wmma(
    const __bf16* __restrict__ hbf_in, const __bf16* __restrict__ Whh,
    const __bf16* __restrict__ Xt,     const __bf16* __restrict__ Wih,
    const float* __restrict__ bias,    float* __restrict__ c,
    float* __restrict__ h, __bf16* __restrict__ hbf_out)
{
    const int lane = threadIdx.x & 31;
    const int wave = threadIdx.x >> 5;
    const int lm   = lane & 15;
    const int half = lane >> 4;
    const int m0   = blockIdx.y * 128 + wave * 16;
    const int n0   = blockIdx.x * 16;

    floatx8 acc[4] = {};
    gemm_acc(hbf_in, Whh, DD,  (size_t)DD * DD,  m0, n0, lm, half, acc);
    gemm_acc(Xt,     Wih, NIN, (size_t)DD * NIN, m0, n0, lm, half, acc);

    const int n = n0 + lm;
    const float bi = bias[n];
    const float bf = bias[DD + n];
    const float bg = bias[2 * DD + n];
    const float bo = bias[3 * DD + n];
#pragma unroll
    for (int r = 0; r < 8; ++r) {
        const int m = m0 + half * 8 + r;
        const size_t idx = (size_t)m * DD + n;
        const float gi = sigm(acc[0][r] + bi);
        const float gf = sigm(acc[1][r] + bf);
        const float gg = tanhf(acc[2][r] + bg);
        const float go = sigm(acc[3][r] + bo);
        const float c2 = gf * c[idx] + gi * gg;
        const float h2 = go * tanhf(c2);
        c[idx] = c2;
        h[idx] = h2;
        hbf_out[idx] = (__bf16)h2;
    }
}

// ---------------------------------------------------------------------------
// Setup kernels
// ---------------------------------------------------------------------------
__global__ void k_conv_bf16(const float* __restrict__ src, __bf16* __restrict__ dst, int n) {
    int i = blockIdx.x * blockDim.x + threadIdx.x;
    if (i < n) dst[i] = (__bf16)src[i];
}

// dst[C,R] = src[R,C]^T  (for W_wa: [D,E] -> Bt rows indexed by e, length D)
__global__ void k_conv_tr_bf16(const float* __restrict__ src, __bf16* __restrict__ dst,
                               int R, int C) {
    int i = blockIdx.x * blockDim.x + threadIdx.x;
    if (i < R * C) {
        int r = i / C, c = i % C;
        dst[(size_t)c * R + r] = (__bf16)src[i];
    }
}

__global__ void k_add_vec(const float* __restrict__ a, const float* __restrict__ b,
                          float* __restrict__ o, int n) {
    int i = blockIdx.x * blockDim.x + threadIdx.x;
    if (i < n) o[i] = a[i] + b[i];
}

// inp0[b, 0:256] = mean_s cEnc[b,s,:]; inp0[b, 256:287] = hist; inp0[b,287] = cur[b,0]
__global__ __launch_bounds__(256) void k_prep_inp0(
    const float* __restrict__ cEnc, const float* __restrict__ hist,
    const float* __restrict__ cur, __bf16* __restrict__ inp0)
{
    const int b = blockIdx.x, e = threadIdx.x;
    const float* p = cEnc + (size_t)b * SS * EE + e;
    float s = 0.f;
    for (int i = 0; i < SS; ++i) s += p[(size_t)i * EE];
    inp0[(size_t)b * (EE + NIN) + e] = (__bf16)(s * (1.0f / SS));
    if (e < HIST) inp0[(size_t)b * (EE + NIN) + EE + e] = (__bf16)hist[(size_t)b * HIST + e];
    if (e == HIST) inp0[(size_t)b * (EE + NIN) + EE + HIST] = (__bf16)cur[(size_t)b * TT + 0];
}

// X[t][b][i] = (t+i < 31) ? hist[b, t+i] : cur[b, t+i-31]
__global__ void k_build_x(const float* __restrict__ hist, const float* __restrict__ cur,
                          __bf16* __restrict__ X) {
    int idx = blockIdx.x * blockDim.x + threadIdx.x;
    if (idx >= TT * BB * NIN) return;
    int t = idx / (BB * NIN);
    int r = idx - t * (BB * NIN);
    int b = r / NIN, i = r - b * NIN;
    int u = t + i;
    float v = (u < HIST) ? hist[(size_t)b * HIST + u] : cur[(size_t)b * TT + (u - HIST)];
    X[idx] = (__bf16)v;
}

// ---------------------------------------------------------------------------
// Fused attention tail (one block of 256 threads per batch row):
//   aligned = S*sigmoid(dot(tanh(Wp h2), Wvp) + bvp)
//   att[s]  = dot(cEnc[b,s,:], v[b,:])           (b_wa drops under softmax)
//   alpha   = softmax(att) * exp(-0.5*(s-aligned)^2)    -> alphas out
//   awe[e]  = sum_s cEnc[b,s,e]*alpha[s]
//   pred    = dot([awe,h2], Wfc) + bfc                  -> preds out
// ---------------------------------------------------------------------------
__global__ __launch_bounds__(256) void k_attn_fused(
    const float* __restrict__ cEnc, const float* __restrict__ v,
    const float* __restrict__ tmp,  const float* __restrict__ Wvp,
    const float* __restrict__ bvp,  const float* __restrict__ h2,
    const float* __restrict__ Wfc,  const float* __restrict__ bfc,
    float* __restrict__ preds, float* __restrict__ alphas, int t)
{
    __shared__ float s_red[256];
    __shared__ float s_att[SS];
    __shared__ float s_alpha[SS];

    const int b = blockIdx.x, tid = threadIdx.x;
    const float* enc_b = cEnc + (size_t)b * SS * EE;

    // ---- aligned position ----
    float p = tmp[(size_t)b * DD + tid] * Wvp[tid]
            + tmp[(size_t)b * DD + 256 + tid] * Wvp[256 + tid];
    s_red[tid] = p; __syncthreads();
    for (int off = 128; off > 0; off >>= 1) {
        if (tid < off) s_red[tid] += s_red[tid + off];
        __syncthreads();
    }
    const float aligned = (float)SS * sigm(s_red[0] + bvp[0]);
    __syncthreads();

    // ---- attention scores: 4 lanes per s ----
    {
        const int s = tid >> 2, q = tid & 3;
        const float* vb = v + (size_t)b * EE;
        float pa = 0.f;
        for (int i = 0; i < EE / 4; ++i) {
            const int e = q + 4 * i;
            pa += enc_b[(size_t)s * EE + e] * vb[e];
        }
        pa += __shfl_xor(pa, 1, 32);
        pa += __shfl_xor(pa, 2, 32);
        if (q == 0) s_att[s] = pa;
    }
    __syncthreads();

    // ---- softmax * gaussian window ----
    float mx = -1e30f;
    for (int s = 0; s < SS; ++s) mx = fmaxf(mx, s_att[s]);
    if (tid < SS) s_alpha[tid] = expf(s_att[tid] - mx);
    __syncthreads();
    float denom = 0.f;
    for (int s = 0; s < SS; ++s) denom += s_alpha[s];
    __syncthreads();
    if (tid < SS) {
        float d0 = (float)tid - aligned;
        float al = (s_alpha[tid] / denom) * expf(-0.5f * d0 * d0);
        s_alpha[tid] = al;
        alphas[((size_t)b * TT + t) * SS + tid] = al;
    }
    __syncthreads();

    // ---- weighted encoding + prediction ----
    float aw = 0.f;
    for (int s = 0; s < SS; ++s) aw += enc_b[(size_t)s * EE + tid] * s_alpha[s];
    float pp = aw * Wfc[tid]
             + h2[(size_t)b * DD + tid]       * Wfc[EE + tid]
             + h2[(size_t)b * DD + 256 + tid] * Wfc[EE + 256 + tid];
    s_red[tid] = pp; __syncthreads();
    for (int off = 128; off > 0; off >>= 1) {
        if (tid < off) s_red[tid] += s_red[tid + off];
        __syncthreads();
    }
    if (tid == 0) preds[(size_t)b * TT + t] = s_red[0] + bfc[0];
}

// ---------------------------------------------------------------------------
extern "C" void kernel_launch(void* const* d_in, const int* in_sizes, int n_in,
                              void* d_out, int out_size, void* d_ws, size_t ws_size,
                              hipStream_t stream) {
    (void)in_sizes; (void)n_in; (void)out_size; (void)ws_size;
    const float* cEnc  = (const float*)d_in[0];
    const float* curSp = (const float*)d_in[1];
    const float* hisSp = (const float*)d_in[2];
    // d_in[3..5]: curAccelXs / histAccelXs / decodeLength — unused by reference math
    const float* Winh = (const float*)d_in[6];
    const float* binh = (const float*)d_in[7];
    const float* Winc = (const float*)d_in[8];
    const float* binc = (const float*)d_in[9];
    const float* Wih  = (const float*)d_in[10];
    const float* Whh  = (const float*)d_in[11];
    const float* bih  = (const float*)d_in[12];
    const float* bhh  = (const float*)d_in[13];
    const float* Wwa  = (const float*)d_in[14];
    // b_wa (d_in[15]) provably drops out under softmax
    const float* Wwp  = (const float*)d_in[16];
    const float* bwp  = (const float*)d_in[17];
    const float* Wvp  = (const float*)d_in[18];
    const float* bvp  = (const float*)d_in[19];
    const float* Wfc  = (const float*)d_in[20];
    const float* bfc  = (const float*)d_in[21];

    float* preds  = (float*)d_out;
    float* alphas = preds + (size_t)BB * TT;

    // ---- workspace arena ----
    char* base = (char*)d_ws;
    size_t off = 0;
    auto alloc = [&](size_t bytes) {
        void* p = base + off;
        off = (off + bytes + 255) & ~(size_t)255;
        return p;
    };
    __bf16* Wih_bf  = (__bf16*)alloc((size_t)G4D * NIN * 2);
    __bf16* Whh_bf  = (__bf16*)alloc((size_t)G4D * DD * 2);
    __bf16* Wwp_bf  = (__bf16*)alloc((size_t)DD * DD * 2);
    __bf16* WwaT_bf = (__bf16*)alloc((size_t)EE * DD * 2);
    __bf16* Winh_bf = (__bf16*)alloc((size_t)DD * (EE + NIN) * 2);
    __bf16* Winc_bf = (__bf16*)alloc((size_t)DD * (EE + NIN) * 2);
    __bf16* inp0    = (__bf16*)alloc((size_t)BB * (EE + NIN) * 2);
    __bf16* Xall    = (__bf16*)alloc((size_t)TT * BB * NIN * 2);
    float*  bias_g  = (float*) alloc((size_t)G4D * 4);
    float*  hbuf    = (float*) alloc((size_t)BB * DD * 4);
    __bf16* hbfA    = (__bf16*)alloc((size_t)BB * DD * 2);
    __bf16* hbfB    = (__bf16*)alloc((size_t)BB * DD * 2);
    float*  cbuf    = (float*) alloc((size_t)BB * DD * 4);
    float*  tmpbuf  = (float*) alloc((size_t)BB * DD * 4);
    float*  vbuf    = (float*) alloc((size_t)BB * EE * 4);

    auto cvt = [&](const float* s, __bf16* d, int n) {
        k_conv_bf16<<<(n + 255) / 256, 256, 0, stream>>>(s, d, n);
    };
    // ---- one-time prep ----
    cvt(Wih,  Wih_bf,  G4D * NIN);
    cvt(Whh,  Whh_bf,  G4D * DD);
    cvt(Wwp,  Wwp_bf,  DD * DD);
    cvt(Winh, Winh_bf, DD * (EE + NIN));
    cvt(Winc, Winc_bf, DD * (EE + NIN));
    k_conv_tr_bf16<<<(DD * EE + 255) / 256, 256, 0, stream>>>(Wwa, WwaT_bf, DD, EE);
    k_add_vec<<<(G4D + 255) / 256, 256, 0, stream>>>(bih, bhh, bias_g, G4D);
    k_prep_inp0<<<BB, 256, 0, stream>>>(cEnc, hisSp, curSp, inp0);
    k_build_x<<<(TT * BB * NIN + 255) / 256, 256, 0, stream>>>(hisSp, curSp, Xall);

    auto gemm = [&](const __bf16* A1, const __bf16* B1, int K1,
                    const float* bias, float* C, __bf16* Cbf, int N, int act) {
        dim3 grid(N / 64, BB / 128);
        gemm_bf16_wmma<<<grid, 256, 0, stream>>>(A1, B1, K1, bias, C, Cbf, N, act);
    };

    // ---- initial hidden / cell state ----
    gemm(inp0, Winh_bf, EE + NIN, binh, hbuf, hbfA, DD, 0);
    gemm(inp0, Winc_bf, EE + NIN, binc, cbuf, nullptr, DD, 0);

    __bf16* h_cur = hbfA;
    __bf16* h_nxt = hbfB;

    // ---- decode loop ----
    for (int t = 0; t < TT; ++t) {
        const __bf16* Xt = Xall + (size_t)t * BB * NIN;
        // fused: gates GEMM (h@Whh^T + x@Wih^T + bias) + LSTM pointwise
        {
            dim3 grid(DD / 16, BB / 128);
            lstm_step_wmma<<<grid, 256, 0, stream>>>(h_cur, Whh_bf, Xt, Wih_bf,
                                                     bias_g, cbuf, hbuf, h_nxt);
        }
        // tmp = tanh(h2 @ Wwp^T + bwp)
        gemm(h_nxt, Wwp_bf, DD, bwp, tmpbuf, nullptr, DD, 1);
        // v = h2 @ Wwa  (Bt rows = Wwa columns)
        gemm(h_nxt, WwaT_bf, DD, nullptr, vbuf, nullptr, EE, 0);
        k_attn_fused<<<BB, 256, 0, stream>>>(cEnc, vbuf, tmpbuf, Wvp, bvp,
                                             hbuf, Wfc, bfc, preds, alphas, t);
        __bf16* sw = h_cur; h_cur = h_nxt; h_nxt = sw;
    }
}
